// WindowAttention3D_44538810859825
// MI455X (gfx1250) — matrix operand order repaired and checked by reading, not compiled, not run on hardware
//
#include <hip/hip_runtime.h>
#include <hip/hip_bf16.h>

// ---------------------------------------------------------------------------
// Fused 3D window attention for gfx1250 (MI455X), bf16 WMMA, f32 accumulate.
// One block per window (2048 blocks), 256 threads = 8 wave32.
// ---------------------------------------------------------------------------

typedef __bf16 bf16_t;
typedef __attribute__((ext_vector_type(16))) __bf16 bf16x16;
typedef __attribute__((ext_vector_type(8)))  float  floatx8;
typedef __attribute__((ext_vector_type(4)))  float  floatx4;
typedef __attribute__((ext_vector_type(4)))  unsigned int uint4v;
typedef __attribute__((ext_vector_type(2)))  unsigned int uint2v;

#define NTOK   98
#define NPAD   112          // 7 * 16
#define KPAD   128          // attn K padded to 4 * 32
#define CDIM   256
#define NHEAD  8
#define HD     32
#define SCALE  0.17677669529663687f   // 1/sqrt(32)

// LDS layout (bytes)
#define OFF_X    0          // 112x256 bf16 (57344) ; reused as 112x112 f32 attn (50176)
#define OFF_Q    57344      // 112x256 bf16
#define OFF_K    114688     // 112x256 bf16
#define OFF_VT   172032     // 256x128 bf16 (V transposed, token dim padded to 128)
#define OFF_P    237568     // 112x128 bf16 (softmaxed attn)
#define OFF_O    266240     // 112x256 bf16
#define SMEM_BYTES 323584   // <= 320KB WGP LDS

__device__ __forceinline__ unsigned short f2bf(float f) {
  unsigned int u = __float_as_uint(f);
  unsigned int r = u + 0x7FFFu + ((u >> 16) & 1u);   // round-to-nearest-even
  return (unsigned short)(r >> 16);
}

union FragU { uint4v q[2]; bf16x16 v; };

// A operand (16x32 bf16): lane holds row (lane&15); lane group g=lane>>4 picks
// K interleave {g*8..g*8+7, 16+g*8..16+g*8+7}  -> two 16B loads.
__device__ __forceinline__ bf16x16 frag_A(const unsigned short* row, int k0, int g) {
  FragU u;
  u.q[0] = *(const uint4v*)(row + k0 + g * 8);
  u.q[1] = *(const uint4v*)(row + k0 + 16 + g * 8);
  return u.v;
}

// B operand (32x16 bf16): lane holds column (lane&15); lane group g picks
// K interleave {g*8..g*8+7, 16+g*8..16+g*8+7} -> two 16B loads from the k-major row.
__device__ __forceinline__ bf16x16 frag_B(const unsigned short* row, int k0, int g) {
  FragU u;
  u.q[0] = *(const uint4v*)(row + k0 + g * 8);
  u.q[1] = *(const uint4v*)(row + k0 + 16 + g * 8);
  return u.v;
}

__device__ __forceinline__ floatx8 wmma_bf16(bf16x16 a, bf16x16 b, floatx8 c) {
  return __builtin_amdgcn_wmma_f32_16x16x32_bf16(false, a, false, b, (short)0, c,
                                                 false, false);
}

// ---------------------------------------------------------------------------
// Prep: cast weights to bf16, pre-gather relative-position bias to (H,98,98).
// ---------------------------------------------------------------------------
__global__ void prep_kernel(const float* __restrict__ qkv_w,
                            const float* __restrict__ proj_w,
                            const float* __restrict__ bias_table,
                            const int*   __restrict__ rel_index,
                            unsigned short* __restrict__ wq,
                            unsigned short* __restrict__ wp,
                            float* __restrict__ biasH) {
  int i = blockIdx.x * blockDim.x + threadIdx.x;
  int stride = gridDim.x * blockDim.x;
  for (int j = i; j < 3 * CDIM * CDIM; j += stride) wq[j] = f2bf(qkv_w[j]);
  for (int j = i; j < CDIM * CDIM; j += stride)     wp[j] = f2bf(proj_w[j]);
  for (int j = i; j < NHEAD * NTOK * NTOK; j += stride) {
    int h = j / (NTOK * NTOK);
    int mn = j % (NTOK * NTOK);
    biasH[j] = bias_table[rel_index[mn] * NHEAD + h];
  }
}

// ---------------------------------------------------------------------------
// Fused attention kernel.
// ---------------------------------------------------------------------------
__global__ __launch_bounds__(256)
void win_attn_fused(const float* __restrict__ x,
                    const float* __restrict__ mask,
                    const float* __restrict__ qkv_b,
                    const float* __restrict__ proj_b,
                    const unsigned short* __restrict__ wq,
                    const unsigned short* __restrict__ wp,
                    const float* __restrict__ biasH,
                    float* __restrict__ out) {
  extern __shared__ char smem[];
  unsigned short* Xs    = (unsigned short*)(smem + OFF_X);
  float*          attnF = (float*)(smem + OFF_X);        // alias after QKV
  unsigned short* Qs    = (unsigned short*)(smem + OFF_Q);
  unsigned short* Ks    = (unsigned short*)(smem + OFF_K);
  unsigned short* VTs   = (unsigned short*)(smem + OFF_VT);
  unsigned short* Ps    = (unsigned short*)(smem + OFF_P);
  unsigned short* Outs  = (unsigned short*)(smem + OFF_O);

  const int tid  = threadIdx.x;
  const int wid  = tid >> 5;       // wave id 0..7
  const int lane = tid & 31;
  const int lg   = lane >> 4;      // lane group
  const int ln   = lane & 15;

  const int b = blockIdx.x;
  const float* xb = x + (size_t)b * NTOK * CDIM;

  // ---- load x -> bf16 LDS, vectorized: b128 global loads, b64 LDS stores ---
  {
    const floatx4* xb4 = (const floatx4*)xb;
    uint2v* xs2 = (uint2v*)Xs;
    for (int i = tid; i < NPAD * CDIM / 4; i += 256) {
      const int m = i >> 6;              // 64 float4 per row
      floatx4 v = {0.f, 0.f, 0.f, 0.f};
      if (m < NTOK) v = xb4[i];
      uint2v p;
      p.x = (unsigned int)f2bf(v.x) | ((unsigned int)f2bf(v.y) << 16);
      p.y = (unsigned int)f2bf(v.z) | ((unsigned int)f2bf(v.w) << 16);
      xs2[i] = p;
    }
    // zero VT (covers the padded token columns; live region overwritten below)
    uint4v z = {0u, 0u, 0u, 0u};
    uint4v* vt4 = (uint4v*)VTs;
    for (int i = tid; i < CDIM * KPAD / 8; i += 256) vt4[i] = z;
  }
  __syncthreads();

  // ---- QKV GEMM: (112x256) @ (256x768)^T tiles; 48 N-tiles / 8 waves -----
  for (int nt = wid; nt < 48; nt += 8) {
    floatx8 acc[7] = {};
    const int c0 = nt * 16;
    const unsigned short* wrow = wq + (size_t)(c0 + ln) * CDIM;
    for (int kt = 0; kt < 8; ++kt) {
      bf16x16 bfrag = frag_B(wrow, kt * 32, lg);
#pragma unroll
      for (int mt = 0; mt < 7; ++mt) {
        bf16x16 afrag = frag_A(Xs + (mt * 16 + ln) * CDIM, kt * 32, lg);
        acc[mt] = wmma_bf16(afrag, bfrag, acc[mt]);
      }
    }
    const float bias = qkv_b[c0 + ln];
#pragma unroll
    for (int mt = 0; mt < 7; ++mt) {
#pragma unroll
      for (int r = 0; r < 8; ++r) {
        const int m = mt * 16 + r + 8 * lg;
        const unsigned short v = f2bf(acc[mt][r] + bias);
        if (c0 < 256)        Qs[m * CDIM + c0 + ln] = v;          // Q row-major
        else if (c0 < 512)   Ks[m * CDIM + (c0 - 256) + ln] = v;  // K row-major
        else                 VTs[(c0 - 512 + ln) * KPAD + m] = v; // V transposed
      }
    }
  }
  __syncthreads();

  const float* maskw = mask + (size_t)(b & 511) * NTOK * NTOK;  // w = b % nW

  // ---- per-head attention ----
  for (int h = 0; h < NHEAD; ++h) {
    // S = Q @ K^T : 7x7 tiles, one K-step (hd = 32)
    for (int t = wid; t < 49; t += 8) {
      const int mt = t / 7, nt2 = t % 7;
      bf16x16 a  = frag_A(Qs + (mt * 16 + ln) * CDIM + h * HD, 0, lg);
      bf16x16 bb = frag_B(Ks + (nt2 * 16 + ln) * CDIM + h * HD, 0, lg);
      floatx8 acc = {};
      acc = wmma_bf16(a, bb, acc);
#pragma unroll
      for (int r = 0; r < 8; ++r)
        attnF[(mt * 16 + r + 8 * lg) * NPAD + nt2 * 16 + ln] = acc[r];
    }
    __syncthreads();

    // logit fuse: all 256 threads, coalesced single read of bias+mask
    {
      const float* bi = biasH + (size_t)h * NTOK * NTOK;
      for (int i = tid; i < NTOK * NTOK; i += 256) {
        const int m = i / NTOK, n = i - m * NTOK;
        attnF[m * NPAD + n] = attnF[m * NPAD + n] * SCALE + bi[i] + maskw[i];
      }
    }
    __syncthreads();

    // per-row softmax (LDS only), bf16 P zero-padded to KPAD
    if (tid < NPAD) {
      const int m = tid;
      unsigned short* dst = Ps + m * KPAD;
      if (m < NTOK) {
        float* sr = attnF + m * NPAD;
        float mx = -1e30f;
        for (int n = 0; n < NTOK; ++n) mx = fmaxf(mx, sr[n]);
        float sum = 0.f;
        for (int n = 0; n < NTOK; ++n) {
          const float e = __expf(sr[n] - mx);
          sum += e;
          sr[n] = e;
        }
        const float inv = 1.0f / sum;
        for (int n = 0; n < NTOK; ++n) dst[n] = f2bf(sr[n] * inv);
        for (int n = NTOK; n < KPAD; ++n) dst[n] = 0;
      } else {
        uint4v z = {0u, 0u, 0u, 0u};
        uint4v* d4 = (uint4v*)dst;
#pragma unroll
        for (int n = 0; n < KPAD / 8; ++n) d4[n] = z;
      }
    }
    __syncthreads();

    // O_h = P @ V^T : 7 M-tiles x 2 N-tiles, 4 K-steps of 32
    for (int t = wid; t < 14; t += 8) {
      const int mt = t >> 1, nt2 = t & 1;
      floatx8 acc = {};
#pragma unroll
      for (int kt = 0; kt < 4; ++kt) {
        bf16x16 a  = frag_A(Ps + (mt * 16 + ln) * KPAD, kt * 32, lg);
        bf16x16 bb = frag_B(VTs + (h * HD + nt2 * 16 + ln) * KPAD, kt * 32, lg);
        acc = wmma_bf16(a, bb, acc);
      }
#pragma unroll
      for (int r = 0; r < 8; ++r)
        Outs[(mt * 16 + r + 8 * lg) * CDIM + h * HD + nt2 * 16 + ln] =
            f2bf(acc[r]);
    }
    __syncthreads();
  }

  // ---- output projection: Out(112x256) @ proj_w^T(256x256) + b -> global --
  float* ob = out + (size_t)b * NTOK * CDIM;
  for (int nt = wid; nt < 16; nt += 8) {
    floatx8 acc[7] = {};
    const unsigned short* wrow = wp + (size_t)(nt * 16 + ln) * CDIM;
    for (int kt = 0; kt < 8; ++kt) {
      bf16x16 bb = frag_B(wrow, kt * 32, lg);
#pragma unroll
      for (int mt = 0; mt < 7; ++mt) {
        bf16x16 a = frag_A(Outs + (mt * 16 + ln) * CDIM, kt * 32, lg);
        acc[mt] = wmma_bf16(a, bb, acc[mt]);
      }
    }
    const float bias = proj_b[nt * 16 + ln];
#pragma unroll
    for (int mt = 0; mt < 7; ++mt) {
#pragma unroll
      for (int r = 0; r < 8; ++r) {
        const int m = mt * 16 + r + 8 * lg;
        if (m < NTOK) ob[m * CDIM + nt * 16 + ln] = acc[mt][r] + bias;
      }
    }
  }
}

// ---------------------------------------------------------------------------
extern "C" void kernel_launch(void* const* d_in, const int* in_sizes, int n_in,
                              void* d_out, int out_size, void* d_ws, size_t ws_size,
                              hipStream_t stream) {
  const float* x          = (const float*)d_in[0];
  const float* mask       = (const float*)d_in[1];
  const float* qkv_w      = (const float*)d_in[2];
  const float* qkv_b      = (const float*)d_in[3];
  const float* proj_w     = (const float*)d_in[4];
  const float* proj_b     = (const float*)d_in[5];
  const float* bias_table = (const float*)d_in[6];
  const int*   rel_index  = (const int*)d_in[7];
  float* out = (float*)d_out;

  unsigned short* wq    = (unsigned short*)d_ws;                       // 393216 B
  unsigned short* wp    = (unsigned short*)((char*)d_ws + 393216);     // 131072 B
  float*          biasH = (float*)((char*)d_ws + 524288);              // 307328 B

  prep_kernel<<<384, 256, 0, stream>>>(qkv_w, proj_w, bias_table, rel_index,
                                       wq, wp, biasH);

  hipFuncSetAttribute((const void*)win_attn_fused,
                      hipFuncAttributeMaxDynamicSharedMemorySize, SMEM_BYTES);
  win_attn_fused<<<2048, 256, SMEM_BYTES, stream>>>(x, mask, qkv_b, proj_b,
                                                    wq, wp, biasH, out);
}
